// RoPEMultiheadAttention_49667001811593
// MI455X (gfx1250) — matrix operand order repaired and checked by reading, not compiled, not run on hardware
//
#include <hip/hip_runtime.h>

// ---------------------------------------------------------------------------
// RoPE Multi-head Attention forward for MI455X (gfx1250), wave32 + WMMA f16.
//   B=2, S=2048, D=1024, H=16, Dh=64
// Matrix math: v_wmma_f32_16x16x32_f16 (f16 operands, f32 accumulate).
// Tile staging: global_load_async_to_lds_b128 (ASYNCcnt DMA) when available,
// double-buffered LDS; all fragment loads are 16B ds_load_b128.
// ---------------------------------------------------------------------------

typedef __attribute__((ext_vector_type(16))) _Float16 v16h;
typedef __attribute__((ext_vector_type(8)))  _Float16 v8h;
typedef __attribute__((ext_vector_type(4)))  _Float16 v4h;
typedef __attribute__((ext_vector_type(8)))  float    v8f;
typedef __attribute__((ext_vector_type(4)))  float    v4f;
typedef __attribute__((ext_vector_type(4)))  int      v4i;

union F16x16 { v16h v; v8h h[2]; };   // one WMMA A/B fragment = 2 x 16B loads

static constexpr int BB = 2;
static constexpr int SS = 2048;
static constexpr int DD = 1024;
static constexpr int HH = 16;
static constexpr int DH = 64;

#define WMMA_F16(a, b, c) \
  __builtin_amdgcn_wmma_f32_16x16x32_f16(false, (a), false, (b), (short)0, (c), false, false)

// --- CDNA5 async global->LDS copy (16B per lane), with sync fallback -------
// Builtin signature (from toolchain diagnostic): (int4 AS1*, int4 AS3*, Imm, Imm)
#if __has_builtin(__builtin_amdgcn_global_load_async_to_lds_b128)
#define HAVE_ASYNC_LDS 1
typedef __attribute__((address_space(1))) v4i* as1_v4i_p;
typedef __attribute__((address_space(3))) v4i* as3_v4i_p;
#define ASYNC_CP16(ldsp, gp)                                                   \
    __builtin_amdgcn_global_load_async_to_lds_b128(                            \
        (as1_v4i_p)(gp), (as3_v4i_p)(ldsp), 0, 0)
#define ASYNC_DRAIN() asm volatile("s_wait_asynccnt 0" ::: "memory")
#else
#define HAVE_ASYNC_LDS 0
#define ASYNC_CP16(ldsp, gp) (*(v8h*)(ldsp) = *(const v8h*)(gp))
#define ASYNC_DRAIN() ((void)0)
#endif

// ===========================================================================
// Kernel: elementwise fp32 -> f16 (vectorized x4)
// ===========================================================================
__global__ __launch_bounds__(256)
void cvt_f16(const float* __restrict__ in, _Float16* __restrict__ out, int n4)
{
    const int i = blockIdx.x * 256 + threadIdx.x;
    if (i < n4) {
        v4f a = ((const v4f*)in)[i];
        v4h h = {(_Float16)a.x, (_Float16)a.y, (_Float16)a.z, (_Float16)a.w};
        ((v4h*)out)[i] = h;
    }
}

// ===========================================================================
// Kernel: tiled transpose + fp32 -> f16:  in (K x N) -> out (N x K)
// ===========================================================================
__global__ __launch_bounds__(256)
void transpose_cvt(const float* __restrict__ in, _Float16* __restrict__ out,
                   int K, int N)
{
    __shared__ _Float16 t[32][33];
    const int n0 = blockIdx.x * 32, k0 = blockIdx.y * 32;
    const int tx = threadIdx.x & 31, ty = threadIdx.x >> 5;   // ty: 0..7
    #pragma unroll
    for (int i = 0; i < 4; ++i)
        t[ty + i * 8][tx] = (_Float16)in[(size_t)(k0 + ty + i * 8) * N + n0 + tx];
    __syncthreads();
    #pragma unroll
    for (int i = 0; i < 4; ++i)
        out[(size_t)(n0 + ty + i * 8) * K + k0 + tx] = t[tx][ty + i * 8];
}

// ===========================================================================
// GEMM: C(MxN) = A(MxK,f16) * Bt(NxK,f16)^T + bias(f32), C fp32.
// Block tile 128x128, 8 waves, wave tile 32x64, K-step 32.
// Double-buffered LDS, async DMA staging, all fragments ds_load_b128.
//   As[buf][m][k], Bs[buf][n][k]: stride 40 halves = 80B (16B aligned rows).
// ===========================================================================
__global__ __launch_bounds__(256)
void gemm_f16wmma(const _Float16* __restrict__ A, const _Float16* __restrict__ Bt,
                  const float* __restrict__ bias, float* __restrict__ C,
                  int M, int N, int K)
{
    __shared__ __align__(32) _Float16 As[2][128][40];   // 2 x 10 KB
    __shared__ __align__(32) _Float16 Bs[2][128][40];   // 2 x 10 KB

    const int tid  = threadIdx.x;
    const int lane = tid & 31;
    const int l16  = lane & 15;
    const int half = lane >> 4;
    const int wave = tid >> 5;
    const int wm   = (wave >> 1) * 32;
    const int wn   = (wave & 1) * 64;
    const int m0   = blockIdx.y * 128;
    const int n0   = blockIdx.x * 128;

    // staging: 256 threads x 2 chunks cover 128 rows x 32 halves per matrix
    const int srow = tid >> 2;              // 0..63 (+64 on second chunk)
    const int scol = (tid & 3) * 8;         // halves: 0, 8, 16, 24

    auto stage = [&](int k0, int buf) {
        #pragma unroll
        for (int r = 0; r < 2; ++r) {
            const int row = srow + r * 64;
            ASYNC_CP16(&As[buf][row][scol], A  + (size_t)(m0 + row) * K + k0 + scol);
            ASYNC_CP16(&Bs[buf][row][scol], Bt + (size_t)(n0 + row) * K + k0 + scol);
        }
    };

    v8f acc[2][4] = {};
    const int nk = K >> 5;
    stage(0, 0);

    for (int i = 0; i < nk; ++i) {
        ASYNC_DRAIN();                       // this wave's DMA done
        __syncthreads();                     // everyone's DMA visible
        if (i + 1 < nk) stage((i + 1) << 5, (i + 1) & 1);   // prefetch next tile

        const int buf = i & 1;
        F16x16 af[2];
        #pragma unroll
        for (int fm = 0; fm < 2; ++fm) {
            const int m  = wm + fm * 16 + l16;
            const int kb = half * 8;
            af[fm].h[0] = *(const v8h*)&As[buf][m][kb];
            af[fm].h[1] = *(const v8h*)&As[buf][m][kb + 16];
        }
        #pragma unroll
        for (int fn = 0; fn < 4; ++fn) {
            F16x16 bf;
            const int n  = wn + fn * 16 + l16;
            const int kb = half * 8;
            bf.h[0] = *(const v8h*)&Bs[buf][n][kb];
            bf.h[1] = *(const v8h*)&Bs[buf][n][kb + 16];
            #pragma unroll
            for (int fm = 0; fm < 2; ++fm)
                acc[fm][fn] = WMMA_F16(af[fm].v, bf.v, acc[fm][fn]);
        }
        __syncthreads();                     // tile fully consumed
    }

    #pragma unroll
    for (int fm = 0; fm < 2; ++fm) {
        #pragma unroll
        for (int fn = 0; fn < 4; ++fn) {
            const int n  = n0 + wn + fn * 16 + l16;
            const float bv = bias[n];
            #pragma unroll
            for (int r = 0; r < 8; ++r) {
                const int m = m0 + wm + fm * 16 + r + half * 8;
                C[(size_t)m * N + n] = acc[fm][fn][r] + bv;
            }
        }
    }
}

// ===========================================================================
// RoPE + split:  Q, K : (B,H,S,Dh) f16     Vt : (B,H,Dh,S) f16
// ===========================================================================
__global__ __launch_bounds__(256)
void rope_split(const float* __restrict__ qkv, const float* __restrict__ cosT,
                const float* __restrict__ sinT, _Float16* __restrict__ Q,
                _Float16* __restrict__ Kh, _Float16* __restrict__ Vt)
{
    const int idx = blockIdx.x * blockDim.x + threadIdx.x;     // B*S*D
    if (idx >= BB * SS * DD) return;
    const int d = idx & (DH - 1);
    const int h = (idx >> 6) & (HH - 1);
    const int s = (idx >> 10) & (SS - 1);
    const int b = idx >> 21;

    const float* row = qkv + (size_t)(b * SS + s) * (3 * DD);
    const float c  = cosT[s * DH + d];
    const float sn = sinT[s * DH + d];
    const int   dp  = (d < 32) ? d + 32 : d - 32;
    const float sgn = (d < 32) ? -1.0f : 1.0f;

    const float q  = row[h * DH + d],       qr = row[h * DH + dp];
    const float k  = row[DD + h * DH + d],  kr = row[DD + h * DH + dp];
    const float v  = row[2 * DD + h * DH + d];

    const size_t bh = (size_t)(b * HH + h);
    Q [bh * SS * DH + (size_t)s * DH + d] = (_Float16)(q * c + sgn * qr * sn);
    Kh[bh * SS * DH + (size_t)s * DH + d] = (_Float16)(k * c + sgn * kr * sn);
    Vt[bh * SS * DH + (size_t)d * SS + s] = (_Float16)v;
}

// ===========================================================================
// Flash attention. grid = (S/256, B*H), 8 waves x 32 queries.
// WG shares (b,h): K/V 64-key blocks staged cooperatively (async DMA):
//   Kst[key][dh] for S = Q K^T      Vst[dh][key] for O = P V
// ctx out: (B, S, H*Dh) f16 (feeds the output-projection GEMM directly).
// ===========================================================================
__global__ __launch_bounds__(256)
void flash_attn(const _Float16* __restrict__ Q, const _Float16* __restrict__ Kh,
                const _Float16* __restrict__ Vt, _Float16* __restrict__ ctx)
{
    __shared__ __align__(32) _Float16 Kst[64][72];      //  9.2 KB
    __shared__ __align__(32) _Float16 Vst[64][72];      //  9.2 KB
    __shared__ __align__(32) _Float16 Pst[8][32][72];   // 36.9 KB (per-wave)

    const int tid  = threadIdx.x;
    const int lane = tid & 31;
    const int l16  = lane & 15;
    const int half = lane >> 4;
    const int wave = tid >> 5;
    const int bh   = blockIdx.y;
    const int b    = bh >> 4, h = bh & 15;
    const int q0   = blockIdx.x * 256 + wave * 32;

    const _Float16* Qp = Q  + (size_t)bh * SS * DH;
    const _Float16* Kp = Kh + (size_t)bh * SS * DH;    // (S, Dh)
    const _Float16* Vp = Vt + (size_t)bh * SS * DH;    // (Dh, S)

    const int srow = tid >> 2;                         // 0..63
    const int scol = (tid & 3) * 16;                   // 0,16,32,48

    // resident Q fragments: 2 M-frags x 2 K-chunks
    F16x16 qf[2][2];
    #pragma unroll
    for (int fm = 0; fm < 2; ++fm) {
        const _Float16* qrow = Qp + (size_t)(q0 + fm * 16 + l16) * DH;
        #pragma unroll
        for (int kc = 0; kc < 2; ++kc) {
            const int kb = kc * 32 + half * 8;
            qf[fm][kc].h[0] = *(const v8h*)(qrow + kb);
            qf[fm][kc].h[1] = *(const v8h*)(qrow + kb + 16);
        }
    }

    v8f   o[2][4] = {};
    float mrow[2][8], lrow[2][8];
    #pragma unroll
    for (int fm = 0; fm < 2; ++fm)
        #pragma unroll
        for (int r = 0; r < 8; ++r) { mrow[fm][r] = -3.0e38f; lrow[fm][r] = 0.0f; }

    const float scale = 0.125f;            // 1/sqrt(64)

    for (int j0 = 0; j0 < SS; j0 += 64) {
        // ------------- cooperative async staging of K / V blocks -------------
        __syncthreads();                   // previous block fully consumed
        {
            const _Float16* kg = Kp + (size_t)(j0 + srow) * DH + scol;
            ASYNC_CP16(&Kst[srow][scol],     kg);
            ASYNC_CP16(&Kst[srow][scol + 8], kg + 8);
            const _Float16* vg = Vp + (size_t)srow * SS + j0 + scol;
            ASYNC_CP16(&Vst[srow][scol],     vg);
            ASYNC_CP16(&Vst[srow][scol + 8], vg + 8);
            ASYNC_DRAIN();
        }
        __syncthreads();

        // ---------------- S = Q K^T ----------------
        v8f sc[2][4] = {};
        #pragma unroll
        for (int kc = 0; kc < 2; ++kc) {
            #pragma unroll
            for (int fn = 0; fn < 4; ++fn) {
                F16x16 bf;
                const int n  = fn * 16 + l16;          // key within block
                const int kb = kc * 32 + half * 8;     // dh chunk
                bf.h[0] = *(const v8h*)&Kst[n][kb];
                bf.h[1] = *(const v8h*)&Kst[n][kb + 16];
                #pragma unroll
                for (int fm = 0; fm < 2; ++fm)
                    sc[fm][fn] = WMMA_F16(qf[fm][kc].v, bf.v, sc[fm][fn]);
            }
        }
        // ---------------- online softmax ----------------
        #pragma unroll
        for (int fm = 0; fm < 2; ++fm) {
            #pragma unroll
            for (int r = 0; r < 8; ++r) {
                float mx = sc[fm][0][r];
                #pragma unroll
                for (int fn = 1; fn < 4; ++fn) mx = fmaxf(mx, sc[fm][fn][r]);
                mx *= scale;
                #pragma unroll
                for (int off = 1; off < 16; off <<= 1)
                    mx = fmaxf(mx, __shfl_xor(mx, off, 32));
                const float mnew  = fmaxf(mrow[fm][r], mx);
                const float alpha = __expf(mrow[fm][r] - mnew);
                float rsum = 0.0f;
                #pragma unroll
                for (int fn = 0; fn < 4; ++fn) {
                    const float p = __expf(sc[fm][fn][r] * scale - mnew);
                    sc[fm][fn][r] = p;
                    rsum += p;
                    o[fm][fn][r] *= alpha;
                }
                #pragma unroll
                for (int off = 1; off < 16; off <<= 1)
                    rsum += __shfl_xor(rsum, off, 32);
                mrow[fm][r] = mnew;
                lrow[fm][r] = lrow[fm][r] * alpha + rsum;
            }
        }
        // ------- P: C-fragment -> per-wave LDS -> A-fragment -------
        #pragma unroll
        for (int fm = 0; fm < 2; ++fm)
            #pragma unroll
            for (int fn = 0; fn < 4; ++fn)
                #pragma unroll
                for (int r = 0; r < 8; ++r)
                    Pst[wave][fm * 16 + r + half * 8][fn * 16 + l16] =
                        (_Float16)sc[fm][fn][r];
        asm volatile("s_wait_dscnt 0" ::: "memory");

        F16x16 pf[2][2];
        #pragma unroll
        for (int fm = 0; fm < 2; ++fm) {
            #pragma unroll
            for (int kc = 0; kc < 2; ++kc) {
                const int kb = kc * 32 + half * 8;
                pf[fm][kc].h[0] = *(const v8h*)&Pst[wave][fm * 16 + l16][kb];
                pf[fm][kc].h[1] = *(const v8h*)&Pst[wave][fm * 16 + l16][kb + 16];
            }
        }
        asm volatile("s_wait_dscnt 0" ::: "memory");

        // ---------------- O += P * V ----------------
        #pragma unroll
        for (int kc = 0; kc < 2; ++kc) {
            #pragma unroll
            for (int fn = 0; fn < 4; ++fn) {
                F16x16 bf;
                const int n  = fn * 16 + l16;          // dh
                const int kb = kc * 32 + half * 8;     // key chunk
                bf.h[0] = *(const v8h*)&Vst[n][kb];
                bf.h[1] = *(const v8h*)&Vst[n][kb + 16];
                #pragma unroll
                for (int fm = 0; fm < 2; ++fm)
                    o[fm][fn] = WMMA_F16(pf[fm][kc].v, bf.v, o[fm][fn]);
            }
        }
    }

    // ---------------- normalize, write f16 context (B, S, H*Dh) ----------------
    #pragma unroll
    for (int fm = 0; fm < 2; ++fm)
        #pragma unroll
        for (int fn = 0; fn < 4; ++fn)
            #pragma unroll
            for (int r = 0; r < 8; ++r) {
                const int qrow = q0 + fm * 16 + r + half * 8;
                ctx[(size_t)(b * SS + qrow) * DD + h * DH + fn * 16 + l16] =
                    (_Float16)(o[fm][fn][r] / lrow[fm][r]);
            }
}

// ===========================================================================
// Launch
// ===========================================================================
extern "C" void kernel_launch(void* const* d_in, const int* in_sizes, int n_in,
                              void* d_out, int out_size, void* d_ws, size_t ws_size,
                              hipStream_t stream)
{
    const float* x     = (const float*)d_in[0];   // (B,S,D)
    const float* cosT  = (const float*)d_in[1];   // (S,Dh)
    const float* sinT  = (const float*)d_in[2];   // (S,Dh)
    const float* W_in  = (const float*)d_in[3];   // (D,3D)
    const float* b_in  = (const float*)d_in[4];   // (3D)
    const float* W_out = (const float*)d_in[5];   // (D,D)
    const float* b_out = (const float*)d_in[6];   // (D)
    float*       out   = (float*)d_out;           // (B,S,D)

    const int M = BB * SS;                        // 4096

    // workspace carve-up (~96 MB)
    char* ws = (char*)d_ws;
    float*    qkv  = (float*)ws;     ws += (size_t)M * 3 * DD * sizeof(float);     // 48 MB
    _Float16* x16  = (_Float16*)ws;  ws += (size_t)M * DD * sizeof(_Float16);      //  8 MB
    _Float16* Wt_i = (_Float16*)ws;  ws += (size_t)3 * DD * DD * sizeof(_Float16); //  6 MB
    _Float16* Wt_o = (_Float16*)ws;  ws += (size_t)DD * DD * sizeof(_Float16);     //  2 MB
    _Float16* Qh   = (_Float16*)ws;  ws += (size_t)M * DD * sizeof(_Float16);      //  8 MB
    _Float16* Kh   = (_Float16*)ws;  ws += (size_t)M * DD * sizeof(_Float16);      //  8 MB
    _Float16* Vt   = (_Float16*)ws;  ws += (size_t)M * DD * sizeof(_Float16);      //  8 MB
    _Float16* ctx  = (_Float16*)ws;  ws += (size_t)M * DD * sizeof(_Float16);      //  8 MB

    // 0) operand prep: x -> f16; weights -> f16 transposed (N,K)
    cvt_f16<<<(M * DD / 4 + 255) / 256, 256, 0, stream>>>(x, x16, M * DD / 4);
    transpose_cvt<<<dim3(3 * DD / 32, DD / 32), 256, 0, stream>>>(W_in, Wt_i, DD, 3 * DD);
    transpose_cvt<<<dim3(DD / 32, DD / 32), 256, 0, stream>>>(W_out, Wt_o, DD, DD);

    // 1) qkv = x @ W_in + b_in          (4096 x 3072 x 1024)
    gemm_f16wmma<<<dim3(3 * DD / 128, M / 128), 256, 0, stream>>>(
        x16, Wt_i, b_in, qkv, M, 3 * DD, DD);

    // 2) RoPE + split into Q, K (row-major) and V (transposed), f16
    rope_split<<<(M * DD) / 256, 256, 0, stream>>>(qkv, cosT, sinT, Qh, Kh, Vt);

    // 3) flash attention -> ctx (f16)
    flash_attn<<<dim3(SS / 256, BB * HH), 256, 0, stream>>>(Qh, Kh, Vt, ctx);

    // 4) out = ctx @ W_out + b_out      (4096 x 1024 x 1024)
    gemm_f16wmma<<<dim3(DD / 128, M / 128), 256, 0, stream>>>(
        ctx, Wt_o, b_out, out, M, DD, DD);
}
